// InvolutionResidualBlock_57604101374674
// MI455X (gfx1250) — matrix operand order repaired and checked by reading, not compiled, not run on hardware
//
#include <hip/hip_runtime.h>
#include <cstdint>

#define CDIM 256
#define HDIM 96
#define WDIM 96
#define HW   (HDIM*WDIM)      /* 9216 */
#define BATCH 4
#define NPIX (BATCH*HW)       /* 36864 */
#define KK 9
#define BN_EPS 1e-5f

typedef __attribute__((ext_vector_type(16))) _Float16 v16h;
typedef __attribute__((ext_vector_type(8)))  _Float16 v8h;
typedef __attribute__((ext_vector_type(8)))  float    v8f;
typedef __attribute__((ext_vector_type(4)))  float    v4f;
typedef __attribute__((ext_vector_type(4)))  unsigned int u32x4;
typedef __attribute__((ext_vector_type(4)))  int      i32x4;
typedef __attribute__((ext_vector_type(8)))  int      i32x8;

// ---- Tensor Data Mover availability / arity detection --------------------
#if defined(__HIP_DEVICE_COMPILE__) && defined(__has_builtin)
#if __has_builtin(__builtin_amdgcn_tensor_load_to_lds)
#define USE_TDM 1
#else
#define USE_TDM 0
#endif
#else
#define USE_TDM 0
#endif

#if defined(__has_include)
#if __has_include(<hip/amd_detail/amd_gfx1250_TDM.h>)
#define TDM_SIX_ARGS 1   /* therock-10.0 headers -> clang-23 6-arg builtin */
#else
#define TDM_SIX_ARGS 0   /* ROCm 7.2 clang-22 5-arg builtin */
#endif
#else
#define TDM_SIX_ARGS 0
#endif

#if USE_TDM
// Issue a 2D TDM tile load: global f32 tile (tile_w x tile_h, row stride
// gstride elems) -> LDS at lds_off. TDM pad feature inserts 1 DWORD of LDS
// padding every (1<<(pad_code+1)) DWORDs -> odd LDS pitch, bank-conflict-free
// column reads. D# layout per CDNA5 ISA.
__device__ __forceinline__ void tdm_load_tile_f32(uint32_t lds_off,
                                                  const float* gaddr,
                                                  uint32_t tile_w,
                                                  uint32_t tile_h,
                                                  uint32_t gstride,
                                                  uint32_t pad_code)
{
    const uint64_t ga = (uint64_t)(uintptr_t)gaddr;
    u32x4 g0;
    g0[0] = 1u;                                     // count=1, user descriptor
    g0[1] = lds_off;                                // lds_addr (bytes)
    g0[2] = (uint32_t)ga;                           // global_addr[31:0]
    g0[3] = (uint32_t)((ga >> 32) & 0x01FFFFFFu)    // global_addr[56:32]
          | (2u << 30);                             // type = 2 ("image")

    i32x8 g1;
    g1[0] = (int)((2u << 16)                        // data_size = 4 bytes
          |       (1u << 20)                        // pad_enable
          |       (pad_code << 22));                // pad_interval; pad_amount=0 (1 DW)
    g1[1] = (int)(tile_w << 16);                    // tensor_dim0[15:0]
    g1[2] = (int)(tile_h << 16);                    // tensor_dim1[15:0]
    g1[3] = (int)(tile_w << 16);                    // tile_dim0
    g1[4] = (int)tile_h;                            // tile_dim1 (tile_dim2 = 0)
    g1[5] = (int)gstride;                           // tensor_dim0_stride[31:0]
    g1[6] = 0;                                      // stride0[47:32] / stride1 lo
    g1[7] = 0;                                      // stride1 hi

    i32x4 z4 = {0, 0, 0, 0};
#if TDM_SIX_ARGS
    i32x8 z8 = {0, 0, 0, 0, 0, 0, 0, 0};
    __builtin_amdgcn_tensor_load_to_lds(g0, g1, z4, z4, z8, 0);
#else
    __builtin_amdgcn_tensor_load_to_lds(g0, g1, z4, z4, 0);
#endif
}
#endif // USE_TDM

// ---------------------------------------------------------------------------
// K1: r[b,o,p] = sum_c w_reduce[o,c] * x[b,c,p]
// WG = 256 threads (8 waves). Tile: 64 o x 64 p, K-step = 32 channels.
// X tiles staged f32 by the TDM, double-buffered (issue i+1, wait cnt<=1 ->
// tile i resident; TDM ops complete in-order per wave). Each wave owns one
// A fragment and two p-subtiles -> 2 WMMAs per K-step.
// ---------------------------------------------------------------------------
#define XPITCH 65
__global__ __launch_bounds__(256)
void inv_reduce_gemm(const float* __restrict__ x,
                     const float* __restrict__ w_reduce,
                     float* __restrict__ r)
{
    __shared__ _Float16 sW[64][40];             // [o_local][c_local]
    __shared__ float    sXf[2][32 * XPITCH + 8];// [c_local][p_local] f32, x2 buf

    const int t     = threadIdx.x;
    const int lane  = t & 31;
    const int wid   = t >> 5;
    const int b     = blockIdx.z;
    const int oBase = blockIdx.y * 64;
    const int pBase = blockIdx.x * 64;

    const int oTile = (wid >> 1) * 16;
    const int pT0   = (wid & 1) * 32;
    const int pT1   = pT0 + 16;
    const int lrow  = lane & 15;
    const int lhalf = lane >> 4;

    v8f acc0 = {};
    v8f acc1 = {};
    int cur = 0;

#if USE_TDM
    if (wid == 0) {
        tdm_load_tile_f32((uint32_t)(uintptr_t)&sXf[0][0],
                          x + ((size_t)(b * CDIM)) * HW + pBase,
                          /*tile_w=*/64, /*tile_h=*/32, HW, /*64 DW=*/5);
    }
#endif

    for (int i = 0; i < 8; ++i) {
        const int c0 = i * 32;
        // ---- W tile: 64 x 32, coalesced float4, cvt to f16 ----
        {
            const int ol = t >> 2;
            const int cg = (t & 3) * 8;
            const float* src = w_reduce + (size_t)(oBase + ol) * CDIM + c0 + cg;
            v4f w0 = *(const v4f*)(src);
            v4f w1 = *(const v4f*)(src + 4);
            #pragma unroll
            for (int q = 0; q < 4; ++q) {
                sW[ol][cg + q]     = (_Float16)w0[q];
                sW[ol][cg + 4 + q] = (_Float16)w1[q];
            }
        }
#if USE_TDM
        if (wid == 0) {
            if (i < 7)
                tdm_load_tile_f32((uint32_t)(uintptr_t)&sXf[cur ^ 1][0],
                                  x + ((size_t)(b * CDIM + c0 + 32)) * HW + pBase,
                                  64, 32, HW, 5);
            if (i < 7) __builtin_amdgcn_s_wait_tensorcnt(1);
            else       __builtin_amdgcn_s_wait_tensorcnt(0);
        }
#else
        {
            const int cl = t >> 3;
            const int pc = (t & 7) * 8;
            const float* src = x + ((size_t)(b * CDIM + c0 + cl)) * HW + pBase + pc;
            v4f v0 = *(const v4f*)src;
            v4f v1 = *(const v4f*)(src + 4);
            #pragma unroll
            for (int q = 0; q < 4; ++q) {
                sXf[cur][cl * XPITCH + pc + q]     = v0[q];
                sXf[cur][cl * XPITCH + pc + 4 + q] = v1[q];
            }
        }
#endif
        __syncthreads();

        // ---- A(16x32)/B(32x16) fragments, both in contiguous k = 16*lhalf + j order ----
        const int kb  = lhalf * 8;    // unused; A also uses kb2: lanes<16 -> K 0-15, >=16 -> K 16-31
        const int kb2 = lhalf * 16;   // B: lanes<16 -> K 0-15, >=16 -> K 16-31
        v8h alo = *(const v8h*)&sW[oTile + lrow][kb2];
        v8h ahi = *(const v8h*)&sW[oTile + lrow][kb2 + 8];
        v16h a, b0, b1;
        #pragma unroll
        for (int j = 0; j < 8; ++j) { a[j] = alo[j]; a[8 + j] = ahi[j]; }
        const float* xb = &sXf[cur][0];
        #pragma unroll
        for (int j = 0; j < 16; ++j) {
            b0[j] = (_Float16)xb[(kb2 + j) * XPITCH + pT0 + lrow];
            b1[j] = (_Float16)xb[(kb2 + j) * XPITCH + pT1 + lrow];
        }
        acc0 = __builtin_amdgcn_wmma_f32_16x16x32_f16(false, a, false, b0,
                                                      (short)0, acc0, false, false);
        acc1 = __builtin_amdgcn_wmma_f32_16x16x32_f16(false, a, false, b1,
                                                      (short)0, acc1, false, false);
        __syncthreads();
        cur ^= 1;
    }

    // ---- store r tiles: D layout M = v + 8*(lane/16), N = lane%16 ----
    #pragma unroll
    for (int v = 0; v < 8; ++v) {
        const int o = oBase + oTile + v + 8 * lhalf;
        const size_t row = ((size_t)(b * CDIM + o)) * HW + pBase;
        r[row + pT0 + lrow] = acc0[v];
        r[row + pT1 + lrow] = acc1[v];
    }
}

// ---------------------------------------------------------------------------
// K2: per-channel batch stats -> fused BN scale/shift
// ---------------------------------------------------------------------------
__global__ __launch_bounds__(256)
void bn_stats(const float* __restrict__ r,
              const float* __restrict__ gamma,
              const float* __restrict__ beta,
              float* __restrict__ scale,
              float* __restrict__ shift)
{
    __shared__ float s1[256];
    __shared__ float s2[256];
    const int o = blockIdx.x;
    const int t = threadIdx.x;
    float sum = 0.0f, sq = 0.0f;
    for (int idx = t; idx < NPIX; idx += 256) {
        const int b = idx / HW;
        const int p = idx - b * HW;
        const float v = r[((size_t)(b * CDIM + o)) * HW + p];
        sum += v; sq += v * v;
    }
    s1[t] = sum; s2[t] = sq;
    __syncthreads();
    for (int st = 128; st > 0; st >>= 1) {
        if (t < st) { s1[t] += s1[t + st]; s2[t] += s2[t + st]; }
        __syncthreads();
    }
    if (t == 0) {
        const float mean = s1[0] * (1.0f / NPIX);
        const float var  = s2[0] * (1.0f / NPIX) - mean * mean;
        const float sc   = gamma[o] * rsqrtf(var + BN_EPS);
        scale[o] = sc;
        shift[o] = beta[o] - mean * sc;
    }
}

// ---------------------------------------------------------------------------
// K3: kern[b,k,p] = sum_c w_span[k,c] * relu(r*scale + shift)
// w_span zero-padded 9->16 rows staged once; r tiles (32 c x 128 p) staged by
// TDM, double-buffered; BN+ReLU fused into B-operand build.
// ---------------------------------------------------------------------------
#define RPITCH 129
__global__ __launch_bounds__(256)
void span_gemm(const float* __restrict__ r,
               const float* __restrict__ w_span,
               const float* __restrict__ scale,
               const float* __restrict__ shift,
               float* __restrict__ kern)
{
    __shared__ _Float16 sA[16][264];             // padded pitch, 16B-aligned rows
    __shared__ float sScale[CDIM];
    __shared__ float sShift[CDIM];
    __shared__ float sR[2][32 * RPITCH + 8];     // [c_local][p_local] f32, x2 buf

    const int t     = threadIdx.x;
    const int lane  = t & 31;
    const int wid   = t >> 5;
    const int b     = blockIdx.z;
    const int lrow  = lane & 15;
    const int lhalf = lane >> 4;
    const int pB    = blockIdx.x * 128;
    const int pl    = wid * 16 + lrow;
    const int p     = pB + pl;

    // stage padded w_span (16 x 256) into LDS, rows >= 9 are zero
    {
        const int row = t >> 4;
        const int cb  = (t & 15) * 16;
        #pragma unroll
        for (int q = 0; q < 16; ++q) {
            const float v = (row < KK) ? w_span[row * CDIM + cb + q] : 0.0f;
            sA[row][cb + q] = (_Float16)v;
        }
    }
    if (t < CDIM) { sScale[t] = scale[t]; sShift[t] = shift[t]; }

    v8f acc = {};
    int cur = 0;

#if USE_TDM
    if (wid == 0) {
        tdm_load_tile_f32((uint32_t)(uintptr_t)&sR[0][0],
                          r + ((size_t)(b * CDIM)) * HW + pB,
                          /*tile_w=*/128, /*tile_h=*/32, HW, /*128 DW=*/6);
    }
#endif

    for (int i = 0; i < 8; ++i) {
        const int c0 = i * 32;
#if USE_TDM
        if (wid == 0) {
            if (i < 7)
                tdm_load_tile_f32((uint32_t)(uintptr_t)&sR[cur ^ 1][0],
                                  r + ((size_t)(b * CDIM + c0 + 32)) * HW + pB,
                                  128, 32, HW, 6);
            if (i < 7) __builtin_amdgcn_s_wait_tensorcnt(1);
            else       __builtin_amdgcn_s_wait_tensorcnt(0);
        }
#else
        {
            const int cl = t >> 3;        // 0..31
            const int pc = (t & 7) * 16;  // 0..112
            const float* src = r + ((size_t)(b * CDIM + c0 + cl)) * HW + pB + pc;
            #pragma unroll
            for (int q = 0; q < 4; ++q) {
                v4f v = *(const v4f*)(src + q * 4);
                #pragma unroll
                for (int e = 0; e < 4; ++e) sR[cur][cl * RPITCH + pc + q * 4 + e] = v[e];
            }
        }
#endif
        __syncthreads();

        const int kb  = lhalf * 8;
        const int kb2 = lhalf * 16;
        v8h alo = *(const v8h*)&sA[lrow][c0 + kb2];
        v8h ahi = *(const v8h*)&sA[lrow][c0 + kb2 + 8];
        v16h a, bb;
        #pragma unroll
        for (int j = 0; j < 8; ++j) { a[j] = alo[j]; a[8 + j] = ahi[j]; }
        const float* rb = &sR[cur][0];
        #pragma unroll
        for (int j = 0; j < 16; ++j) {
            const int c = c0 + kb2 + j;
            float v = rb[(kb2 + j) * RPITCH + pl] * sScale[c] + sShift[c];
            bb[j] = (_Float16)fmaxf(v, 0.0f);
        }
        acc = __builtin_amdgcn_wmma_f32_16x16x32_f16(false, a, false, bb,
                                                     (short)0, acc, false, false);
        __syncthreads();
        cur ^= 1;
    }

    #pragma unroll
    for (int v = 0; v < 8; ++v) {
        const int k = v + 8 * lhalf;
        if (k < KK)
            kern[((size_t)(b * KK + k)) * HW + p] = acc[v];
    }
}

// ---------------------------------------------------------------------------
// K4: out[b,c,h,w] = 2 * sum_{k} x[b,c,h+dh,w+dw] * kern[b,k,h,w]
// kern[9] shared by all 256 channels -> L2 reuse; memory-bound gather.
// ---------------------------------------------------------------------------
__global__ __launch_bounds__(256)
void involution_apply(const float* __restrict__ x,
                      const float* __restrict__ kern,
                      float* __restrict__ out)
{
    const int idx = blockIdx.x * 256 + threadIdx.x;
    const int w = idx % WDIM;
    const int h = (idx / WDIM) % HDIM;
    const int c = (idx / HW) % CDIM;
    const int b = idx / (HW * CDIM);

    const float* kb = kern + ((size_t)b * KK) * HW + h * WDIM + w;
    const float* xb = x + ((size_t)(b * CDIM + c)) * HW;

    float accv = 0.0f;
    #pragma unroll
    for (int dh = -1; dh <= 1; ++dh) {
        #pragma unroll
        for (int dw = -1; dw <= 1; ++dw) {
            const int k  = (dh + 1) * 3 + (dw + 1);
            const int hh = h + dh, ww = w + dw;
            if (hh >= 0 && hh < HDIM && ww >= 0 && ww < WDIM)
                accv += xb[hh * WDIM + ww] * kb[(size_t)k * HW];
        }
    }
    out[idx] = 2.0f * accv;
}

// ---------------------------------------------------------------------------
extern "C" void kernel_launch(void* const* d_in, const int* in_sizes, int n_in,
                              void* d_out, int out_size, void* d_ws, size_t ws_size,
                              hipStream_t stream)
{
    const float* x        = (const float*)d_in[0];
    const float* w_reduce = (const float*)d_in[1];
    const float* w_span   = (const float*)d_in[2];
    const float* gamma    = (const float*)d_in[3];
    const float* beta     = (const float*)d_in[4];
    float* out = (float*)d_out;

    // workspace layout (floats): r | scale | shift | kern
    float* ws    = (float*)d_ws;
    float* r     = ws;                                  // 4*256*9216 = 9,437,184
    float* scale = r + (size_t)BATCH * CDIM * HW;       // 256
    float* shift = scale + CDIM;                        // 256
    float* kern  = shift + CDIM;                        // 4*9*9216 = 331,776

    inv_reduce_gemm<<<dim3(HW / 64, CDIM / 64, BATCH), 256, 0, stream>>>(x, w_reduce, r);
    bn_stats<<<dim3(CDIM), 256, 0, stream>>>(r, gamma, beta, scale, shift);
    span_gemm<<<dim3(HW / 128, 1, BATCH), 256, 0, stream>>>(r, w_span, scale, shift, kern);
    involution_apply<<<dim3((BATCH * CDIM * HW) / 256), 256, 0, stream>>>(x, kern, out);
}
